// Attention_71657234366656
// MI455X (gfx1250) — matrix operand order repaired and checked by reading, not compiled, not run on hardware
//
#include <hip/hip_runtime.h>
#include <math.h>

// Problem constants (from reference setup_inputs)
#define NBATCH 8
#define LQ     2048
#define LKK    2048
#define DHEAD  512
#define SCALE  0.04419417382415922f   // 1/sqrt(512)

typedef __attribute__((ext_vector_type(16))) __bf16 v16bf;
typedef __attribute__((ext_vector_type(4)))  __bf16 v4bf;
typedef __attribute__((ext_vector_type(8)))  float  v8f;

// D = A*B + C, fp32 accumulate, bf16 inputs
#define WMMA_BF16(acc, a, b) \
  (acc) = __builtin_amdgcn_wmma_f32_16x16x32_bf16(false, (a), false, (b), (short)0, (acc), false, false)

__device__ __forceinline__ v8f zero_v8f() {
  v8f z;
#pragma unroll
  for (int i = 0; i < 8; ++i) z[i] = 0.0f;
  return z;
}

// Split 16 fp32 values into bf16 hi + bf16 lo (3-product fp32 emulation,
// error ~2^-17 relative).
__device__ __forceinline__ void split_bf16(const float f[16], v16bf &hi, v16bf &lo) {
#pragma unroll
  for (int i = 0; i < 16; ++i) {
    __bf16 h = (__bf16)f[i];
    hi[i] = h;
    lo[i] = (__bf16)(f[i] - (float)h);
  }
}

// A-operand loader (16x32 bf16, row-major f32 source, K contiguous).
// Lane L<16: M=L, elements 0-15 -> K=c+0..15.
// Lane L>=16: M=L-16, elements 0-15 -> K=c+16..31.
__device__ __forceinline__ void load_a_split(const float *__restrict__ rowp, int c,
                                             int lane, v16bf &hi, v16bf &lo) {
  const int o = (lane < 16) ? 0 : 16;
  float f[16];
  const float4 x0 = *(const float4 *)(rowp + c + o);
  const float4 x1 = *(const float4 *)(rowp + c + o + 4);
  const float4 x2 = *(const float4 *)(rowp + c + o + 8);
  const float4 x3 = *(const float4 *)(rowp + c + o + 12);
  f[0]=x0.x; f[1]=x0.y; f[2]=x0.z;  f[3]=x0.w;
  f[4]=x1.x; f[5]=x1.y; f[6]=x1.z;  f[7]=x1.w;
  f[8]=x2.x; f[9]=x2.y; f[10]=x2.z; f[11]=x2.w;
  f[12]=x3.x; f[13]=x3.y; f[14]=x3.z; f[15]=x3.w;
  split_bf16(f, hi, lo);
}

// B-operand loader from pre-split bf16 (16 contiguous bf16 = 32 B, aligned).
__device__ __forceinline__ v16bf load_b16(const __bf16 *__restrict__ p) {
  return *(const v16bf *)p;
}

// f32 B-operand loader for fallback path (NT case, contiguous).
__device__ __forceinline__ void load_b_contig_split(const float *__restrict__ p,
                                                    v16bf &hi, v16bf &lo) {
  float f[16];
  const float4 x0 = *(const float4 *)(p + 0);
  const float4 x1 = *(const float4 *)(p + 4);
  const float4 x2 = *(const float4 *)(p + 8);
  const float4 x3 = *(const float4 *)(p + 12);
  f[0]=x0.x; f[1]=x0.y; f[2]=x0.z;  f[3]=x0.w;
  f[4]=x1.x; f[5]=x1.y; f[6]=x1.z;  f[7]=x1.w;
  f[8]=x2.x; f[9]=x2.y; f[10]=x2.z; f[11]=x2.w;
  f[12]=x3.x; f[13]=x3.y; f[14]=x3.z; f[15]=x3.w;
  split_bf16(f, hi, lo);
}

// ---------------------------------------------------------------------------
// Pass 0a: split K (f32, [b][k][d]) into bf16 hi/lo, same layout.
// ---------------------------------------------------------------------------
__global__ __launch_bounds__(256) void k_convert_kernel(const float *__restrict__ src,
                                                        __bf16 *__restrict__ hi,
                                                        __bf16 *__restrict__ lo) {
  const size_t i = (size_t)blockIdx.x * 256 + threadIdx.x;   // float4 index
  const float4 x = ((const float4 *)src)[i];
  float f[4] = {x.x, x.y, x.z, x.w};
  v4bf h, l;
#pragma unroll
  for (int j = 0; j < 4; ++j) {
    __bf16 hj = (__bf16)f[j];
    h[j] = hj;
    l[j] = (__bf16)(f[j] - (float)hj);
  }
  *(v4bf *)(hi + 4 * i) = h;
  *(v4bf *)(lo + 4 * i) = l;
}

// ---------------------------------------------------------------------------
// Pass 0b: split + transpose V: f32 [b][k][d] -> bf16 hi/lo [b][d][k].
// 32x32 LDS tile transpose; block (32,8), 4 rows per thread.
// ---------------------------------------------------------------------------
__global__ __launch_bounds__(256) void vt_convert_kernel(const float *__restrict__ V,
                                                         __bf16 *__restrict__ Vthi,
                                                         __bf16 *__restrict__ Vtlo) {
  __shared__ float tile[32][33];
  const int b  = blockIdx.z;
  const int d0 = blockIdx.x * 32;
  const int k0 = blockIdx.y * 32;
  const int tx = threadIdx.x;
  const int ty = threadIdx.y;
  const float *vb = V + (size_t)b * LKK * DHEAD;
#pragma unroll
  for (int i = 0; i < 4; ++i) {
    const int kl = ty + i * 8;
    tile[kl][tx] = vb[(size_t)(k0 + kl) * DHEAD + d0 + tx];
  }
  __syncthreads();
  __bf16 *hib = Vthi + (size_t)b * DHEAD * LKK;
  __bf16 *lob = Vtlo + (size_t)b * DHEAD * LKK;
#pragma unroll
  for (int i = 0; i < 4; ++i) {
    const int dl = ty + i * 8;
    const float x = tile[tx][dl];
    const __bf16 h = (__bf16)x;
    hib[(size_t)(d0 + dl) * LKK + k0 + tx] = h;
    lob[(size_t)(d0 + dl) * LKK + k0 + tx] = (__bf16)(x - (float)h);
  }
}

// ---------------------------------------------------------------------------
// Pass 1 (fast): S = Q * K^T with pre-split K.
// Workgroup macro tile: 128(M) x 128(N), 8 waves.
// Wave w: M sub (w&3)*32 (two 16-row A operands), N sub (w>>2)*64 (4 tiles).
// Every B tile (hi+lo) feeds 6 WMMAs -> 24 WMMA per 24 b128 loads per chunk.
// ---------------------------------------------------------------------------
__global__ __launch_bounds__(256) void qk_kernel_pre(const float *__restrict__ Q,
                                                     const __bf16 *__restrict__ Khi,
                                                     const __bf16 *__restrict__ Klo,
                                                     float *__restrict__ W) {
  const int lane  = threadIdx.x & 31;
  const int wave  = threadIdx.x >> 5;
  const int b     = blockIdx.z;
  const int qbase = blockIdx.y * 128 + (wave & 3) * 32;
  const int nbase = blockIdx.x * 128 + (wave >> 2) * 64;
  const int lrow  = lane & 15;
  const int boff  = (lane < 16) ? 0 : 16;

  const float *qrow0 = Q + ((size_t)b * LQ + qbase + lrow) * DHEAD;
  const float *qrow1 = qrow0 + (size_t)16 * DHEAD;
  const __bf16 *khp[4], *klp[4];
#pragma unroll
  for (int t = 0; t < 4; ++t) {
    const size_t r = ((size_t)b * LKK + nbase + t * 16 + lrow) * DHEAD + boff;
    khp[t] = Khi + r;
    klp[t] = Klo + r;
  }

  v8f acc[2][4];
#pragma unroll
  for (int mi = 0; mi < 2; ++mi)
#pragma unroll
    for (int t = 0; t < 4; ++t) acc[mi][t] = zero_v8f();

  for (int c = 0; c < DHEAD; c += 32) {
    v16bf ahi0, alo0, ahi1, alo1;
    load_a_split(qrow0, c, lane, ahi0, alo0);
    load_a_split(qrow1, c, lane, ahi1, alo1);
#pragma unroll
    for (int t = 0; t < 4; ++t) {
      const v16bf bhi = load_b16(khp[t] + c);
      const v16bf blo = load_b16(klp[t] + c);
      WMMA_BF16(acc[0][t], ahi0, bhi);
      WMMA_BF16(acc[0][t], alo0, bhi);
      WMMA_BF16(acc[0][t], ahi0, blo);
      WMMA_BF16(acc[1][t], ahi1, bhi);
      WMMA_BF16(acc[1][t], alo1, bhi);
      WMMA_BF16(acc[1][t], ahi1, blo);
    }
  }

  float *wout = W + (size_t)b * LQ * LKK;
#pragma unroll
  for (int mi = 0; mi < 2; ++mi) {
#pragma unroll
    for (int t = 0; t < 4; ++t) {
      const int col = nbase + t * 16 + lrow;
#pragma unroll
      for (int r = 0; r < 8; ++r) {
        const int m = qbase + mi * 16 + ((lane < 16) ? r : r + 8);
        wout[(size_t)m * LKK + col] = acc[mi][t][r];
      }
    }
  }
}

// ---------------------------------------------------------------------------
// Pass 3 (fast): O = W * V with pre-split, pre-transposed V ([b][d][k]).
// Same 128x128 macro tile / 32x64 wave tile as pass 1.
// ---------------------------------------------------------------------------
__global__ __launch_bounds__(256) void wv_kernel_pre(const float *__restrict__ W,
                                                     const __bf16 *__restrict__ Vthi,
                                                     const __bf16 *__restrict__ Vtlo,
                                                     float *__restrict__ O) {
  const int lane  = threadIdx.x & 31;
  const int wave  = threadIdx.x >> 5;
  const int b     = blockIdx.z;
  const int qbase = blockIdx.y * 128 + (wave & 3) * 32;
  const int nbase = blockIdx.x * 128 + (wave >> 2) * 64;
  const int lrow  = lane & 15;
  const int boff  = (lane < 16) ? 0 : 16;

  const float *wrow0 = W + ((size_t)b * LQ + qbase + lrow) * LKK;
  const float *wrow1 = wrow0 + (size_t)16 * LKK;
  const __bf16 *vhp[4], *vlp[4];
#pragma unroll
  for (int t = 0; t < 4; ++t) {
    const size_t r = ((size_t)b * DHEAD + nbase + t * 16 + lrow) * LKK + boff;
    vhp[t] = Vthi + r;
    vlp[t] = Vtlo + r;
  }

  v8f acc[2][4];
#pragma unroll
  for (int mi = 0; mi < 2; ++mi)
#pragma unroll
    for (int t = 0; t < 4; ++t) acc[mi][t] = zero_v8f();

  for (int c = 0; c < LKK; c += 32) {
    v16bf ahi0, alo0, ahi1, alo1;
    load_a_split(wrow0, c, lane, ahi0, alo0);
    load_a_split(wrow1, c, lane, ahi1, alo1);
#pragma unroll
    for (int t = 0; t < 4; ++t) {
      const v16bf bhi = load_b16(vhp[t] + c);
      const v16bf blo = load_b16(vlp[t] + c);
      WMMA_BF16(acc[0][t], ahi0, bhi);
      WMMA_BF16(acc[0][t], alo0, bhi);
      WMMA_BF16(acc[0][t], ahi0, blo);
      WMMA_BF16(acc[1][t], ahi1, bhi);
      WMMA_BF16(acc[1][t], alo1, bhi);
      WMMA_BF16(acc[1][t], ahi1, blo);
    }
  }

#pragma unroll
  for (int mi = 0; mi < 2; ++mi) {
#pragma unroll
    for (int t = 0; t < 4; ++t) {
      const int col = nbase + t * 16 + lrow;
#pragma unroll
      for (int r = 0; r < 8; ++r) {
        const int m = qbase + mi * 16 + ((lane < 16) ? r : r + 8);
        O[((size_t)b * LQ + m) * DHEAD + col] = acc[mi][t][r];
      }
    }
  }
}

// ---------------------------------------------------------------------------
// Pass 2: row-wise masked, scaled, numerically-stable softmax in place.
// All-masked rows (rowmax == -inf) produce 0.0 (reference NaN->0 rule).
// ---------------------------------------------------------------------------
__global__ __launch_bounds__(256) void softmax_kernel(float *__restrict__ W,
                                                      const unsigned char *__restrict__ mask) {
  const int row = blockIdx.x;   // 0 .. NBATCH*LQ-1
  const int b   = row / LQ;
  float *w = W + (size_t)row * LKK;
  const unsigned char *mrow = mask + (size_t)b * LKK;
  const int tid = threadIdx.x;

  __shared__ float red[256];

  float v[8];
  float m = -INFINITY;
#pragma unroll
  for (int i = 0; i < 8; ++i) {
    const int k = tid + i * 256;
    float s = w[k] * SCALE;
    s = mrow[k] ? -INFINITY : s;
    v[i] = s;
    m = fmaxf(m, s);
  }
  red[tid] = m;
  __syncthreads();
  for (int off = 128; off > 0; off >>= 1) {
    if (tid < off) red[tid] = fmaxf(red[tid], red[tid + off]);
    __syncthreads();
  }
  m = red[0];
  __syncthreads();

  const bool dead = !(m > -INFINITY);
  float sum = 0.0f;
#pragma unroll
  for (int i = 0; i < 8; ++i) {
    const float e = dead ? 0.0f : expf(v[i] - m);
    v[i] = e;
    sum += e;
  }
  red[tid] = sum;
  __syncthreads();
  for (int off = 128; off > 0; off >>= 1) {
    if (tid < off) red[tid] += red[tid + off];
    __syncthreads();
  }
  sum = red[0];

  const float inv = dead ? 0.0f : (1.0f / sum);
#pragma unroll
  for (int i = 0; i < 8; ++i) {
    const int k = tid + i * 256;
    w[k] = v[i] * inv;
  }
}

// ---------------------------------------------------------------------------
// Fallback GEMMs (no workspace): on-the-fly f32 -> bf16 hi/lo split.
// ---------------------------------------------------------------------------
__global__ __launch_bounds__(256) void qk_kernel_fb(const float *__restrict__ Q,
                                                    const float *__restrict__ K,
                                                    float *__restrict__ W) {
  const int lane  = threadIdx.x & 31;
  const int wave  = threadIdx.x >> 5;
  const int b     = blockIdx.z;
  const int qbase = blockIdx.y * 64 + (wave & 3) * 16;
  const int nbase = blockIdx.x * 128 + (wave >> 2) * 64;
  const int lrow  = lane & 15;

  const float *qrow = Q + ((size_t)b * LQ + qbase + lrow) * DHEAD;
  const float *kmat = K + (size_t)b * LKK * DHEAD;

  v8f acc[4];
#pragma unroll
  for (int t = 0; t < 4; ++t) acc[t] = zero_v8f();

  for (int c = 0; c < DHEAD; c += 32) {
    v16bf ahi, alo;
    load_a_split(qrow, c, lane, ahi, alo);
#pragma unroll
    for (int t = 0; t < 4; ++t) {
      const float *kp = kmat + (size_t)(nbase + t * 16 + lrow) * DHEAD + c +
                        ((lane < 16) ? 0 : 16);
      v16bf bhi, blo;
      load_b_contig_split(kp, bhi, blo);
      WMMA_BF16(acc[t], ahi, bhi);
      WMMA_BF16(acc[t], alo, bhi);
      WMMA_BF16(acc[t], ahi, blo);
    }
  }

  float *wout = W + (size_t)b * LQ * LKK;
#pragma unroll
  for (int t = 0; t < 4; ++t) {
    const int col = nbase + t * 16 + lrow;
#pragma unroll
    for (int r = 0; r < 8; ++r) {
      const int m = qbase + ((lane < 16) ? r : r + 8);
      wout[(size_t)m * LKK + col] = acc[t][r];
    }
  }
}

__global__ __launch_bounds__(256) void wv_kernel_fb(const float *__restrict__ W,
                                                    const float *__restrict__ V,
                                                    float *__restrict__ O) {
  const int lane  = threadIdx.x & 31;
  const int wave  = threadIdx.x >> 5;
  const int b     = blockIdx.z;
  const int qbase = blockIdx.y * 64 + (wave & 3) * 16;
  const int nbase = blockIdx.x * 128 + (wave >> 2) * 64;
  const int lrow  = lane & 15;

  const float *wrow = W + ((size_t)b * LQ + qbase + lrow) * LKK;
  const float *vmat = V + (size_t)b * LKK * DHEAD;

  v8f acc[4];
#pragma unroll
  for (int t = 0; t < 4; ++t) acc[t] = zero_v8f();

  for (int c = 0; c < LKK; c += 32) {
    v16bf ahi, alo;
    load_a_split(wrow, c, lane, ahi, alo);
    const int koff = c + ((lane < 16) ? 0 : 16);
#pragma unroll
    for (int t = 0; t < 4; ++t) {
      const int col = nbase + t * 16 + lrow;
      float f[16];
#pragma unroll
      for (int e = 0; e < 16; ++e)
        f[e] = vmat[(size_t)(koff + e) * DHEAD + col];
      v16bf bhi, blo;
      split_bf16(f, bhi, blo);
      WMMA_BF16(acc[t], ahi, bhi);
      WMMA_BF16(acc[t], alo, bhi);
      WMMA_BF16(acc[t], ahi, blo);
    }
  }

#pragma unroll
  for (int t = 0; t < 4; ++t) {
    const int col = nbase + t * 16 + lrow;
#pragma unroll
    for (int r = 0; r < 8; ++r) {
      const int m = qbase + ((lane < 16) ? r : r + 8);
      O[((size_t)b * LQ + m) * DHEAD + col] = acc[t][r];
    }
  }
}

extern "C" void kernel_launch(void *const *d_in, const int *in_sizes, int n_in,
                              void *d_out, int out_size, void *d_ws, size_t ws_size,
                              hipStream_t stream) {
  (void)in_sizes; (void)n_in; (void)out_size;

  const float *Q = (const float *)d_in[0];
  const float *K = (const float *)d_in[1];
  const float *V = (const float *)d_in[2];
  const unsigned char *mask = (const unsigned char *)d_in[3];  // jax bool_: 1 byte

  float *out_vals = (float *)d_out;                                   // (B, Lq, D)
  float *out_w    = out_vals + (size_t)NBATCH * LQ * DHEAD;           // (B, Lq, Lk)

  const size_t NELEM   = (size_t)NBATCH * LKK * DHEAD;                // 8.39 M
  const size_t WS_NEED = 4 * NELEM * sizeof(__bf16);                  // 64 MiB

  if (ws_size >= WS_NEED) {
    __bf16 *Khi  = (__bf16 *)d_ws;
    __bf16 *Klo  = Khi + NELEM;
    __bf16 *Vthi = Klo + NELEM;
    __bf16 *Vtlo = Vthi + NELEM;

    // Pass 0: one-time bf16 hi/lo split of K, and split+transpose of V.
    k_convert_kernel<<<dim3((unsigned)(NELEM / 4 / 256)), dim3(256), 0, stream>>>(K, Khi, Klo);
    vt_convert_kernel<<<dim3(DHEAD / 32, LKK / 32, NBATCH), dim3(32, 8), 0, stream>>>(V, Vthi, Vtlo);
    // Pass 1: raw scores into the weights output region (used as scratch).
    qk_kernel_pre<<<dim3(LKK / 128, LQ / 128, NBATCH), dim3(256), 0, stream>>>(Q, Khi, Klo, out_w);
    // Pass 2: masked, scaled softmax in place -> final weights output.
    softmax_kernel<<<dim3(NBATCH * LQ), dim3(256), 0, stream>>>(out_w, mask);
    // Pass 3: O = W * V -> values output.
    wv_kernel_pre<<<dim3(DHEAD / 128, LQ / 128, NBATCH), dim3(256), 0, stream>>>(out_w, Vthi, Vtlo, out_vals);
  } else {
    // Workspace too small: on-the-fly conversion path.
    qk_kernel_fb<<<dim3(LKK / 128, LQ / 64, NBATCH), dim3(256), 0, stream>>>(Q, K, out_w);
    softmax_kernel<<<dim3(NBATCH * LQ), dim3(256), 0, stream>>>(out_w, mask);
    wv_kernel_fb<<<dim3(DHEAD / 128, LQ / 64, NBATCH), dim3(256), 0, stream>>>(out_w, V, out_vals);
  }
}
